// FeedForward_Network_33887291966167
// MI455X (gfx1250) — matrix operand order repaired and checked by reading, not compiled, not run on hardware
//
#include <hip/hip_runtime.h>

typedef __attribute__((ext_vector_type(16))) _Float16 v16h;
typedef __attribute__((ext_vector_type(8)))  _Float16 v8h;
typedef __attribute__((ext_vector_type(8)))  float    v8f;
typedef __attribute__((ext_vector_type(4)))  float    v4f;

#define LDS_STRIDE 20   // floats per row: 16B-aligned rows, conflict-free stores

template <bool GUARD>
__global__ __launch_bounds__(256) void mlp_sim_wmma(
    const float* __restrict__ x,
    const float* __restrict__ W1, const float* __restrict__ b1,
    const float* __restrict__ W2, const float* __restrict__ b2,
    const float* __restrict__ W3, const float* __restrict__ b3,
    float* __restrict__ out, int n)
{
  __shared__ __align__(16) float lds[8][2][16][LDS_STRIDE];

  const int lane = threadIdx.x & 31;
  const int wave = threadIdx.x >> 5;
  const int hi   = lane >> 4;     // A: K-half / D: M-half this lane covers
  const int col  = lane & 15;     // A: row m / B,C,D: column n

  // ---------------- loop-invariant operand setup ----------------
  const int i0 = hi * 8;          // this lane produces h1[i0..i0+7]
  v8h w1h, b1h;
#pragma unroll
  for (int j = 0; j < 8; ++j) {
    w1h[j] = (_Float16)W1[i0 + j];
    b1h[j] = (_Float16)b1[i0 + j];
  }

  // B = W2^T (32x16 f16 layout, table order): lane n, half h holds
  // B[k=8h+s, n] = W2[n][8h+s] for s=0..7; elements 8..15 (K=16..31) -> zero pad.
  v16h bmat = {};
  {
#pragma unroll
    for (int k = 0; k < 8; ++k)
      bmat[k] = (_Float16)W2[col * 16 + i0 + k];
  }

  // C = b2 broadcast: C[m, n=col] = b2[col] for all 8 accumulator rows
  v8f cbias;
  {
    const float bb = b2[col];
#pragma unroll
    for (int r = 0; r < 8; ++r) cbias[r] = bb;
  }

  // W3 / b3 are wave-uniform -> SGPRs
  float w3s[16];
#pragma unroll
  for (int k = 0; k < 16; ++k) w3s[k] = W3[k];
  const float bias3 = b3[0];

  const int nWaves = (int)(gridDim.x * (blockDim.x >> 5));
  const int waveId = (int)(blockIdx.x * (blockDim.x >> 5)) + wave;
  const int stride = nWaves * 32;

  // h1 = relu(x*W1 + b1), packed f16, laid out for the WMMA A operand
  auto makeA = [&](float xv) -> v16h {
    const _Float16 xf = (_Float16)xv;
    v8h xh;
#pragma unroll
    for (int j = 0; j < 8; ++j) xh[j] = xf;
    v8h h1 = xh * w1h + b1h;                    // v_pk_fma_f16
#if __has_builtin(__builtin_elementwise_max)
    v8h z = {};
    h1 = __builtin_elementwise_max(h1, z);      // v_pk_max_num_f16
#else
#pragma unroll
    for (int j = 0; j < 8; ++j)
      h1[j] = h1[j] > (_Float16)0 ? h1[j] : (_Float16)0;
#endif
    v16h a = {};                                // K=16..31 zero-padded
#pragma unroll
    for (int j = 0; j < 8; ++j) a[j] = h1[j];
    return a;
  };

  // Process 32 consecutive elements (2 independent WMMA tiles).
  auto body = [&](int base, float xv0, float xv1) {
    const v16h a0 = makeA(xv0);
    const v16h a1 = makeA(xv1);
    // two independent WMMAs back-to-back (second fills hazard slots of first)
    const v8f d0 = __builtin_amdgcn_wmma_f32_16x16x32_f16(
        false, a0, false, bmat, (short)0, cbias, false, false);
    const v8f d1 = __builtin_amdgcn_wmma_f32_16x16x32_f16(
        false, a1, false, bmat, (short)0, cbias, false, false);

    // single-instruction relu (v_med3_f32) + transpose-scatter to LDS
#pragma unroll
    for (int r = 0; r < 8; ++r) {
      lds[wave][0][i0 + r][col] = __builtin_amdgcn_fmed3f(d0[r], 0.0f, 1.0e30f);
      lds[wave][1][i0 + r][col] = __builtin_amdgcn_fmed3f(d1[r], 0.0f, 1.0e30f);
    }

    // same-wave DS ops are in-order; enforce store->load visibility anyway
    __asm__ volatile("s_wait_dscnt 0" ::: "memory");

    // Gather: lane l owns element base+l -> tile hi, row col (16 contiguous,
    // 16B-aligned floats -> 4x ds_load_b128).
    const v4f* rowv = (const v4f*)&lds[wave][hi][col][0];
    const v4f r0 = rowv[0], r1 = rowv[1], r2 = rowv[2], r3 = rowv[3];

    float acc = bias3;
#pragma unroll
    for (int k = 0; k < 4; ++k) acc = __builtin_fmaf(w3s[k],      r0[k], acc);
#pragma unroll
    for (int k = 0; k < 4; ++k) acc = __builtin_fmaf(w3s[4 + k],  r1[k], acc);
#pragma unroll
    for (int k = 0; k < 4; ++k) acc = __builtin_fmaf(w3s[8 + k],  r2[k], acc);
#pragma unroll
    for (int k = 0; k < 4; ++k) acc = __builtin_fmaf(w3s[12 + k], r3[k], acc);

    // 1 - sigmoid(acc) = 1 / (1 + exp(acc))
    const float res = __builtin_amdgcn_rcpf(1.0f + __expf(acc));

    const int e = base + lane;
    if (!GUARD || e < n) __builtin_nontemporal_store(res, &out[e]);

    __asm__ volatile("" ::: "memory");  // keep next iter's DS stores after loads
  };

  if (GUARD) {
    for (int base = waveId * 32; base < n; base += stride) {
      const int e0 = base + col, e1 = base + 16 + col;
      float xv0 = 0.0f, xv1 = 0.0f;
      if (e0 < n) xv0 = __builtin_nontemporal_load(&x[e0]);
      if (e1 < n) xv1 = __builtin_nontemporal_load(&x[e1]);
      body(base, xv0, xv1);
    }
  } else {
    // n % 32 == 0: no per-lane guards, x loads software-pipelined 1 iter ahead
    int base = waveId * 32;
    if (base < n) {
      float xv0 = __builtin_nontemporal_load(&x[base + col]);
      float xv1 = __builtin_nontemporal_load(&x[base + 16 + col]);
      for (;;) {
        const int nb = base + stride;
        float nx0 = 0.0f, nx1 = 0.0f;
        if (nb < n) {  // wave-uniform branch, no EXEC churn
          nx0 = __builtin_nontemporal_load(&x[nb + col]);
          nx1 = __builtin_nontemporal_load(&x[nb + 16 + col]);
        }
        body(base, xv0, xv1);
        if (nb >= n) break;
        base = nb; xv0 = nx0; xv1 = nx1;
      }
    }
  }
}

extern "C" void kernel_launch(void* const* d_in, const int* in_sizes, int n_in,
                              void* d_out, int out_size, void* d_ws, size_t ws_size,
                              hipStream_t stream) {
  const float* x  = (const float*)d_in[0];
  const float* W1 = (const float*)d_in[1];
  const float* b1 = (const float*)d_in[2];
  const float* W2 = (const float*)d_in[3];
  const float* b2 = (const float*)d_in[4];
  const float* W3 = (const float*)d_in[5];
  const float* b3 = (const float*)d_in[6];
  float* out = (float*)d_out;

  const int n = in_sizes[0];                 // Q*B elements
  int waves  = (n + 31) / 32;
  int blocks = (waves + 7) / 8;              // 8 waves (256 threads) per block
  if (blocks > 4096) blocks = 4096;
  if (blocks < 1) blocks = 1;

  if ((n & 31) == 0) {
    hipLaunchKernelGGL(mlp_sim_wmma<false>, dim3(blocks), dim3(256), 0, stream,
                       x, W1, b1, W2, b2, W3, b3, out, n);
  } else {
    hipLaunchKernelGGL(mlp_sim_wmma<true>, dim3(blocks), dim3(256), 0, stream,
                       x, W1, b1, W2, b2, W3, b3, out, n);
  }
}
